// SpatialGraphEncoder_63694365000320
// MI455X (gfx1250) — matrix operand order repaired and checked by reading, not compiled, not run on hardware
//
#include <hip/hip_runtime.h>
#include <math.h>

#define N_NODES 50000
#define N_EDGES 800000
#define E_TOT   850000          // edges + self loops
#define HID     128
#define EDIM    64
#define NEG_SLOPE 0.2f
#define WE_STRIDE 72            // padded LDS stride (halfs): 16B-aligned, conflict-free

typedef __attribute__((ext_vector_type(16))) _Float16 v16h;
typedef __attribute__((ext_vector_type(8)))  _Float16 v8h;
typedef __attribute__((ext_vector_type(8)))  float    v8f;

__device__ inline void atomicMaxFloat(float* addr, float val) {
    int* ai = (int*)addr;
    int cur = __float_as_int(*addr);
    while (__int_as_float(cur) < val) {
        int prev = atomicCAS(ai, cur, __float_as_int(val));
        if (prev == cur) break;
        cur = prev;
    }
}

__device__ inline v16h join16(v8h lo, v8h hi) {
    return __builtin_shufflevector(lo, hi, 0,1,2,3,4,5,6,7,8,9,10,11,12,13,14,15);
}

// ---------------------------------------------------------------------------
// Node transform: Xl = A@Wl + bl, Xr = A@Wr + br
// A: f16 [50000, K] row-major; Wt: f16 [128, K] (transposed weights)
// grid.x = 50000/16 row tiles, grid.y = 2 (left/right), 8 waves = 8 col tiles
// ---------------------------------------------------------------------------
__global__ void __launch_bounds__(256)
node_gemm_wmma(const _Float16* __restrict__ A, int K,
               const _Float16* __restrict__ WtL, const float* __restrict__ bl,
               const _Float16* __restrict__ WtR, const float* __restrict__ br,
               float* __restrict__ Xl, float* __restrict__ Xr)
{
    const int wave = threadIdx.x >> 5;
    const int lane = threadIdx.x & 31;
    const int ln   = lane & 15;
    const int kgrp = lane >> 4;      // 0: lanes 0-15, 1: lanes 16-31
    const int rt   = blockIdx.x;

    const _Float16* Wt   = blockIdx.y ? WtR : WtL;
    const float*    bias = blockIdx.y ? br : bl;
    float*          Out  = blockIdx.y ? Xr : Xl;

    const int row = rt * 16 + ln;
    const int col = wave * 16 + ln;
    const _Float16* arow = A  + (long)row * K;
    const _Float16* wcol = Wt + (long)col * K;

    v8f acc = {0.f,0.f,0.f,0.f,0.f,0.f,0.f,0.f};
    for (int kk = 0; kk < K; kk += 32) {
        const int ka = kk + kgrp * 8;      // A: elems 0-7 -> K=ka.., 8-15 -> K=ka+16..
        v8h alo = *(const v8h*)(arow + ka);
        v8h ahi = *(const v8h*)(arow + ka + 16);
        const int kb = kk + kgrp * 8;      // B: elems 0-7 -> K=kb.., 8-15 -> K=kb+16..
        v8h blo = *(const v8h*)(wcol + kb);
        v8h bhi = *(const v8h*)(wcol + kb + 16);
        acc = __builtin_amdgcn_wmma_f32_16x16x32_f16(
                  false, join16(alo, ahi), false, join16(blo, bhi),
                  (short)0, acc, false, false);
    }
    const float bv = bias[col];
    #pragma unroll
    for (int r = 0; r < 8; ++r) {
        const int m = rt * 16 + kgrp * 8 + r;   // D: VGPR r -> M = r (+8 hi half)
        Out[(long)m * HID + col] = acc[r] + bv;
    }
}

// ---------------------------------------------------------------------------
// Fused edge kernel: he = ea@We (WMMA, We staged transposed f16 in LDS),
// z = leaky_relu(he + xl[src] + xr[dst]), logit = z . att ;
// store logit, atomicMax segment max.  1 wave = 16 edges.
// ---------------------------------------------------------------------------
__global__ void __launch_bounds__(256)
edge_logits_wmma(const float* __restrict__ ea, const float* __restrict__ ea_mean,
                 const long long* __restrict__ eidx,
                 const float* __restrict__ We, const float* __restrict__ att,
                 const float* __restrict__ Xl, const float* __restrict__ Xr,
                 float* __restrict__ logit, float* __restrict__ mbuf)
{
    __shared__ __align__(16) _Float16 Wes[HID * WE_STRIDE];   // We^T, f16, padded

    // cooperative stage: Wes[c][k] = (f16) We[k][c]
    for (int idx = threadIdx.x; idx < HID * EDIM; idx += 256) {
        const int c = idx >> 6;          // 0..127
        const int k = idx & 63;          // 0..63
        Wes[c * WE_STRIDE + k] = (_Float16)We[k * HID + c];
    }
    __syncthreads();

    const int NT   = E_TOT / 16;
    const int tile = blockIdx.x * (blockDim.x >> 5) + (threadIdx.x >> 5);
    if (tile >= NT) return;                 // wave-uniform: EXEC all-ones for WMMA
    const int lane = threadIdx.x & 31;
    const int ln   = lane & 15;
    const int kgrp = lane >> 4;
    const long long* dstI = eidx + N_EDGES;

    const int erow = tile * 16 + ln;        // edge supplying this lane's A rows
    const float4* arow4 = (const float4*)((erow < N_EDGES) ? (ea + (long)erow * EDIM)
                                                           : ea_mean);

    v8f accs[8];
    const v8f vzero = {0.f,0.f,0.f,0.f,0.f,0.f,0.f,0.f};
    #pragma unroll
    for (int ct = 0; ct < 8; ++ct) accs[ct] = vzero;

    #pragma unroll
    for (int kk = 0; kk < EDIM; kk += 32) {
        const int ka4 = (kk >> 2) + kgrp * 2;       // float4 index of K=kk+kgrp*8
        float4 l0 = arow4[ka4];
        float4 l1 = arow4[ka4 + 1];
        float4 h0 = arow4[ka4 + 4];
        float4 h1 = arow4[ka4 + 5];
        v16h a;
        a[0]=(_Float16)l0.x; a[1]=(_Float16)l0.y; a[2]=(_Float16)l0.z; a[3]=(_Float16)l0.w;
        a[4]=(_Float16)l1.x; a[5]=(_Float16)l1.y; a[6]=(_Float16)l1.z; a[7]=(_Float16)l1.w;
        a[8]=(_Float16)h0.x; a[9]=(_Float16)h0.y; a[10]=(_Float16)h0.z; a[11]=(_Float16)h0.w;
        a[12]=(_Float16)h1.x; a[13]=(_Float16)h1.y; a[14]=(_Float16)h1.z; a[15]=(_Float16)h1.w;

        const int kb = kk + kgrp * 8;
        #pragma unroll
        for (int ct = 0; ct < 8; ++ct) {
            const _Float16* wp = &Wes[(ct * 16 + ln) * WE_STRIDE + kb];
            v8h blo = *(const v8h*)wp;
            v8h bhi = *(const v8h*)(wp + 16);
            accs[ct] = __builtin_amdgcn_wmma_f32_16x16x32_f16(
                           false, a, false, join16(blo, bhi),
                           (short)0, accs[ct], false, false);
        }
    }

    // hoist att gathers (loop-invariant per lane)
    float attv[8];
    #pragma unroll
    for (int ct = 0; ct < 8; ++ct) attv[ct] = att[ct * 16 + ln];

    // per-lane partial dot with att over this lane's 8 columns, per edge r
    float part[8];
    #pragma unroll
    for (int r = 0; r < 8; ++r) {
        const int e = tile * 16 + kgrp * 8 + r;
        int s, d;
        if (e < N_EDGES) { s = (int)eidx[e]; d = (int)dstI[e]; }
        else             { s = d = e - N_EDGES; }
        const float* xls = Xl + (long)s * HID;
        const float* xrd = Xr + (long)d * HID;
        float sum = 0.f;
        #pragma unroll
        for (int ct = 0; ct < 8; ++ct) {
            const int c = ct * 16 + ln;
            float z = accs[ct][r] + xls[c] + xrd[c];
            z = (z > 0.f) ? z : NEG_SLOPE * z;
            sum += z * attv[ct];
        }
        part[r] = sum;
    }
    // butterfly reduction within each 16-lane half (halves own disjoint edge sets)
    #pragma unroll
    for (int mask = 1; mask < 16; mask <<= 1) {
        #pragma unroll
        for (int r = 0; r < 8; ++r)
            part[r] += __shfl_xor(part[r], mask, 32);
    }
    #pragma unroll
    for (int r = 0; r < 8; ++r) {
        if (ln == r) {
            const int e = tile * 16 + kgrp * 8 + r;
            logit[e] = part[r];
            const int d = (e < N_EDGES) ? (int)dstI[e] : (e - N_EDGES);
            atomicMaxFloat(&mbuf[d], part[r]);
        }
    }
}

// ---------------------------------------------------------------------------
__global__ void edge_softmax(const float* __restrict__ logit,
                             const long long* __restrict__ eidx,
                             const float* __restrict__ mbuf,
                             float* __restrict__ pbuf, float* __restrict__ sbuf)
{
    const int e = blockIdx.x * blockDim.x + threadIdx.x;
    if (e >= E_TOT) return;
    const int d = (e < N_EDGES) ? (int)eidx[N_EDGES + e] : (e - N_EDGES);
    const float pv = __expf(logit[e] - mbuf[d]);
    pbuf[e] = pv;
    atomicAdd(&sbuf[d], pv);
}

__global__ void edge_aggregate(const float* __restrict__ pbuf, const float* __restrict__ sbuf,
                               const long long* __restrict__ eidx,
                               const float* __restrict__ Xl, float* __restrict__ out)
{
    const int gid  = blockIdx.x * blockDim.x + threadIdx.x;
    const int e    = gid >> 5;
    const int lane = gid & 31;
    if (e >= E_TOT) return;
    int s, d;
    if (e < N_EDGES) { s = (int)eidx[e]; d = (int)eidx[N_EDGES + e]; }
    else             { s = d = e - N_EDGES; }
    const float alpha = pbuf[e] / sbuf[d];
    #pragma unroll
    for (int i = 0; i < 4; ++i) {
        const int c = lane + 32 * i;
        atomicAdd(&out[(long)d * HID + c], alpha * Xl[(long)s * HID + c]);
    }
}

// ---------------------------------------------------------------------------
__global__ void ea_mean_kernel(const float* __restrict__ ea, float* __restrict__ mean)
{
    __shared__ float red[256];
    const int c = blockIdx.x;
    float sum = 0.f;
    for (int e = threadIdx.x; e < N_EDGES; e += 256)
        sum += ea[(long)e * EDIM + c];
    red[threadIdx.x] = sum;
    __syncthreads();
    for (int off = 128; off > 0; off >>= 1) {
        if (threadIdx.x < off) red[threadIdx.x] += red[threadIdx.x + off];
        __syncthreads();
    }
    if (threadIdx.x == 0) mean[c] = red[0] / (float)N_EDGES;
}

__global__ void f32_to_f16(const float* __restrict__ in, _Float16* __restrict__ out, int n)
{
    const int i = blockIdx.x * blockDim.x + threadIdx.x;
    if (i < n) out[i] = (_Float16)in[i];
}

// W: [K, 128] f32  ->  Wt: [128, K] f16
__global__ void transpose_w_f16(const float* __restrict__ W, _Float16* __restrict__ Wt, int K)
{
    const int idx = blockIdx.x * blockDim.x + threadIdx.x;
    if (idx < K * HID) {
        const int c = idx / K;
        const int k = idx - c * K;
        Wt[idx] = (_Float16)W[k * HID + c];
    }
}

__global__ void init_ms(float* __restrict__ m, float* __restrict__ s)
{
    const int i = blockIdx.x * blockDim.x + threadIdx.x;
    if (i < N_NODES) { m[i] = -INFINITY; s[i] = 0.f; }
}

__global__ void init_out_bias(float* __restrict__ out, const float* __restrict__ bias)
{
    const int i = blockIdx.x * blockDim.x + threadIdx.x;
    if (i < N_NODES * HID) out[i] = bias[i & (HID - 1)];
}

// silu in place on f32 h, plus emit f16 copy for next layer's GEMM
__global__ void silu_store(float* __restrict__ h, _Float16* __restrict__ hh)
{
    const int i = blockIdx.x * blockDim.x + threadIdx.x;
    if (i < N_NODES * HID) {
        const float x = h[i];
        const float y = x / (1.f + __expf(-x));
        h[i]  = y;
        hh[i] = (_Float16)y;
    }
}

// ---------------------------------------------------------------------------
extern "C" void kernel_launch(void* const* d_in, const int* in_sizes, int n_in,
                              void* d_out, int out_size, void* d_ws, size_t ws_size,
                              hipStream_t stream)
{
    const float*     x    = (const float*)d_in[0];
    const long long* eidx = (const long long*)d_in[1];   // [2, 800000] int64
    const float*     ea   = (const float*)d_in[2];
    const float* Wl0 = (const float*)d_in[3];
    const float* bl0 = (const float*)d_in[4];
    const float* Wr0 = (const float*)d_in[5];
    const float* br0 = (const float*)d_in[6];
    const float* We0 = (const float*)d_in[7];
    const float* att0= (const float*)d_in[8];
    const float* b0  = (const float*)d_in[9];
    const float* Wl1 = (const float*)d_in[10];
    const float* bl1 = (const float*)d_in[11];
    const float* Wr1 = (const float*)d_in[12];
    const float* br1 = (const float*)d_in[13];
    const float* We1 = (const float*)d_in[14];
    const float* att1= (const float*)d_in[15];
    const float* b1  = (const float*)d_in[16];

    float* ws     = (float*)d_ws;
    float* Xl     = ws;                                  // 6.4M f32
    float* Xr     = Xl + (size_t)N_NODES * HID;          // 6.4M f32
    float* Acc    = Xr + (size_t)N_NODES * HID;          // 6.4M f32
    float* logit  = Acc + (size_t)N_NODES * HID;         // 850K f32
    float* pbuf   = logit + E_TOT;                       // 850K f32
    float* mbuf   = pbuf + E_TOT;                        // 50K f32
    float* sbuf   = mbuf + N_NODES;                      // 50K f32
    float* eamean = sbuf + N_NODES;                      // 64 f32
    _Float16* xh  = (_Float16*)(eamean + 64);            // 12.8M f16
    _Float16* hh  = xh + (size_t)N_NODES * 256;          // 6.4M f16
    _Float16* WtL = hh + (size_t)N_NODES * HID;          // 32K f16
    _Float16* WtR = WtL + 256 * HID;                     // 32K f16

    float* hout = (float*)d_out;

    const int NT   = E_TOT / 16;                 // 53125 edge tiles
    const int gNT  = (NT + 7) / 8;
    const int gN   = (N_NODES + 255) / 256;
    const int gNH  = (N_NODES * HID + 255) / 256;
    const int gE   = (E_TOT + 255) / 256;
    const int gE32 = (E_TOT * 32 + 255) / 256;
    const int gW0  = (256 * HID + 255) / 256;
    const int gW1  = (HID * HID + 255) / 256;

    ea_mean_kernel<<<EDIM, 256, 0, stream>>>(ea, eamean);
    f32_to_f16<<<(N_NODES * 256 + 255) / 256, 256, 0, stream>>>(x, xh, N_NODES * 256);

    // ---------------- layer 0 ----------------
    transpose_w_f16<<<gW0, 256, 0, stream>>>(Wl0, WtL, 256);
    transpose_w_f16<<<gW0, 256, 0, stream>>>(Wr0, WtR, 256);
    node_gemm_wmma<<<dim3(N_NODES / 16, 2), 256, 0, stream>>>(xh, 256, WtL, bl0, WtR, br0, Xl, Xr);
    init_ms<<<gN, 256, 0, stream>>>(mbuf, sbuf);
    init_out_bias<<<gNH, 256, 0, stream>>>(hout, b0);
    edge_logits_wmma<<<gNT, 256, 0, stream>>>(ea, eamean, eidx, We0, att0, Xl, Xr, logit, mbuf);
    edge_softmax<<<gE, 256, 0, stream>>>(logit, eidx, mbuf, pbuf, sbuf);
    edge_aggregate<<<gE32, 256, 0, stream>>>(pbuf, sbuf, eidx, Xl, hout);
    silu_store<<<gNH, 256, 0, stream>>>(hout, hh);

    // ---------------- layer 1 ----------------
    transpose_w_f16<<<gW1, 256, 0, stream>>>(Wl1, WtL, HID);
    transpose_w_f16<<<gW1, 256, 0, stream>>>(Wr1, WtR, HID);
    node_gemm_wmma<<<dim3(N_NODES / 16, 2), 256, 0, stream>>>(hh, 128, WtL, bl1, WtR, br1, Xl, Xr);
    init_ms<<<gN, 256, 0, stream>>>(mbuf, sbuf);
    init_out_bias<<<gNH, 256, 0, stream>>>(Acc, b1);
    edge_logits_wmma<<<gNT, 256, 0, stream>>>(ea, eamean, eidx, We1, att1, Xl, Xr, logit, mbuf);
    edge_softmax<<<gE, 256, 0, stream>>>(logit, eidx, mbuf, pbuf, sbuf);
    edge_aggregate<<<gE32, 256, 0, stream>>>(pbuf, sbuf, eidx, Xl, Acc);

    hipMemcpyAsync(d_out, Acc, (size_t)N_NODES * HID * sizeof(float),
                   hipMemcpyDeviceToDevice, stream);
}
